// Graph_RelNet_6691559047522
// MI455X (gfx1250) — matrix operand order repaired and checked by reading, not compiled, not run on hardware
//
#include <hip/hip_runtime.h>
#include <hip/hip_bf16.h>
#include <math.h>

// ---------------- problem constants ----------------
#define NND   10000      // nodes
#define NED   100000     // edges
#define NGR   16         // graphs
#define EDIM  2049       // raw edge feature dim
#define FDIM  2053       // EDIM + unit(3) + logdist(1)
#define KPAD  2080       // FDIM padded to 65*32 for WMMA K-chunks
#define NKC   65         // KPAD/32
#define HB    128        // hidden width

#define EFULL 1562       // full 64-edge blocks (1562*64 = 99968)
#define NFULL 156        // full 64-node blocks (156*64 = 9984)

// ---------------- vector types for WMMA ----------------
typedef __attribute__((ext_vector_type(16))) __bf16       v16bf;
typedef __attribute__((ext_vector_type(8)))  __bf16       v8bf;
typedef __attribute__((ext_vector_type(8)))  float        v8f;
typedef __attribute__((ext_vector_type(4)))  unsigned int v4u;

union BF16x16 { v16bf v; v8bf h[2]; };

// f32 -> bf16 round-to-nearest-even
__device__ inline unsigned f2bf(float f) {
  unsigned u = __float_as_uint(f);
  u += 0x7fffu + ((u >> 16) & 1u);
  return u >> 16;
}
__device__ inline unsigned pk2bf(float a, float b) {
  return f2bf(a) | (f2bf(b) << 16);
}

// stage a 32x128 bf16 weight chunk (8KB) global -> LDS, 128 threads, b128 ops
__device__ inline void stageB(const unsigned short* __restrict__ W,
                              unsigned short* __restrict__ sB, int tid) {
  const v4u* s = reinterpret_cast<const v4u*>(W);
  v4u*       d = reinterpret_cast<v4u*>(sB);
#pragma unroll
  for (int i = 0; i < 4; ++i) d[tid + i * 128] = s[tid + i * 128];
}

// 8 N-tiles of v_wmma_f32_16x16x32_bf16; B fragment: lane L holds col N=L&15,
// K = kb..kb+7 and kb+16..kb+23 where kb = (L>>4)*8 (matches A)
__device__ inline void wmma_8tiles(v8f acc[8], const BF16x16& a,
                                   const unsigned short* __restrict__ sBc, int lane) {
  const unsigned short* bp = sBc + (lane & 15) + ((lane >> 4) * 8) * HB;
#pragma unroll
  for (int t = 0; t < 8; ++t) {
    BF16x16 b;
#pragma unroll
    for (int s = 0; s < 16; ++s)
      b.v[s] = __builtin_bit_cast(__bf16, bp[(s < 8 ? s : s + 8) * HB + t * 16]);
    acc[t] = __builtin_amdgcn_wmma_f32_16x16x32_bf16(
        false, a.v, false, b.v, (short)0, acc[t], false, false);
  }
}

// C/D layout: value r of acc[t] in lane L = (row M=r+8*(L>>4), col N=16t+(L&15))
__device__ inline void add_bias_relu(v8f acc[8], const float* __restrict__ b, int lane) {
  const int c0 = lane & 15;
#pragma unroll
  for (int t = 0; t < 8; ++t) {
    float bb = b[t * 16 + c0];
#pragma unroll
    for (int r = 0; r < 8; ++r) {
      float v = acc[t][r] + bb;
      acc[t][r] = v > 0.f ? v : 0.f;
    }
  }
}

// LayerNorm over 128 features per row: each row lives in one 16-lane half,
// 8 values (one per N-tile) per lane -> shfl_xor 1/2/4/8 reduces the half.
__device__ inline void ln_rows(v8f acc[8], const float* __restrict__ g,
                               const float* __restrict__ be, int lane) {
  const int c0 = lane & 15;
  float gg[8], bb[8];
#pragma unroll
  for (int t = 0; t < 8; ++t) { gg[t] = g[t * 16 + c0]; bb[t] = be[t * 16 + c0]; }
#pragma unroll
  for (int r = 0; r < 8; ++r) {
    float s = 0.f;
#pragma unroll
    for (int t = 0; t < 8; ++t) s += acc[t][r];
    s += __shfl_xor(s, 1, 32); s += __shfl_xor(s, 2, 32);
    s += __shfl_xor(s, 4, 32); s += __shfl_xor(s, 8, 32);
    float mean = s * (1.0f / 128.0f);
    float s2 = 0.f;
#pragma unroll
    for (int t = 0; t < 8; ++t) { float d = acc[t][r] - mean; s2 += d * d; }
    s2 += __shfl_xor(s2, 1, 32); s2 += __shfl_xor(s2, 2, 32);
    s2 += __shfl_xor(s2, 4, 32); s2 += __shfl_xor(s2, 8, 32);
    float inv = rsqrtf(s2 * (1.0f / 128.0f) + 1e-5f);
#pragma unroll
    for (int t = 0; t < 8; ++t)
      acc[t][r] = (acc[t][r] - mean) * inv * gg[t] + bb[t];
  }
}

// write wave's 16x128 activation tile to LDS as bf16 (row-major, stride 128)
__device__ inline void store_sH(unsigned short* __restrict__ sH,
                                const v8f acc[8], int wv, int lane) {
  const int c0 = lane & 15, kh = lane >> 4;
#pragma unroll
  for (int t = 0; t < 8; ++t)
#pragma unroll
    for (int r = 0; r < 8; ++r)
      sH[(wv * 16 + r + kh * 8) * HB + t * 16 + c0] = (unsigned short)f2bf(acc[t][r]);
}

// A-fragment (16x32 bf16): lane L -> row (L&15), two 8-elem runs at
// K = kb and kb+16 where kb = (L>>4)*8
__device__ inline BF16x16 load_afrag(const unsigned short* __restrict__ base,
                                     int lane, int stride) {
  const int rowA = lane & 15, kb = (lane >> 4) * 8;
  const unsigned short* ap = base + rowA * stride + kb;
  BF16x16 a;
  a.h[0] = *reinterpret_cast<const v8bf*>(ap);
  a.h[1] = *reinterpret_cast<const v8bf*>(ap + 16);
  return a;
}

// 16x128 @ 128x128 matmul, A from per-wave LDS tile, B double-buffered
__device__ inline void mm_sH(v8f acc[8], const unsigned short* __restrict__ sH,
                             const unsigned short* __restrict__ W,
                             unsigned short (*__restrict__ sB)[32 * HB],
                             int wv, int lane, int tid) {
  const v8f zz = {0.f, 0.f, 0.f, 0.f, 0.f, 0.f, 0.f, 0.f};
#pragma unroll
  for (int t = 0; t < 8; ++t) acc[t] = zz;
  __syncthreads();                       // previous users of sB done
  stageB(W, sB[0], tid);
#pragma unroll
  for (int kc = 0; kc < 4; ++kc) {
    __syncthreads();                     // chunk kc staged by all threads
    if (kc < 3) stageB(W + (size_t)(kc + 1) * 32 * HB, sB[(kc + 1) & 1], tid);
    BF16x16 a = load_afrag(sH + wv * 16 * HB + kc * 32, lane, HB);
    wmma_8tiles(acc, a, sB[kc & 1], lane);
  }
}

// ---- A staging for layer 1: packed-pair b32 stores, branch-free when FULL ----
template <bool FULL>
__device__ inline void stageA_main(unsigned short* __restrict__ dst,
                                   const float* __restrict__ eattr,
                                   int e0, int kc, int tid) {
#pragma unroll
  for (int i = 0; i < 8; ++i) {
    int p = tid + i * 128;               // pair index 0..1023
    int el = p >> 4;                     // edge row 0..63
    int k0 = (p & 15) * 2;               // even col 0..30
    float v0 = 0.f, v1 = 0.f;
    if (FULL || (e0 + el) < NED) {
      const float* src = eattr + (size_t)(e0 + el) * EDIM + kc * 32 + k0;
      v0 = src[0];
      v1 = src[1];
    }
    *reinterpret_cast<unsigned*>(dst + el * 32 + k0) = pk2bf(v0, v1);
  }
}

__device__ inline float last_feat(const float* __restrict__ eattr,
                                  const float (*__restrict__ lext)[4],
                                  int e0, int el, int k) {
  if (k == 0) {
    int e = e0 + el;
    return (e < NED) ? eattr[(size_t)e * EDIM + 2048] : 0.f;
  }
  if (k <= 4) return lext[el][k - 1];
  return 0.f;
}

__device__ inline void stageA_last(unsigned short* __restrict__ dst,
                                   const float* __restrict__ eattr,
                                   const float (*__restrict__ lext)[4],
                                   int e0, int tid) {
#pragma unroll
  for (int i = 0; i < 8; ++i) {
    int p = tid + i * 128;
    int el = p >> 4;
    int k0 = (p & 15) * 2;
    float v0 = last_feat(eattr, lext, e0, el, k0);
    float v1 = last_feat(eattr, lext, e0, el, k0 + 1);
    *reinterpret_cast<unsigned*>(dst + el * 32 + k0) = pk2bf(v0, v1);
  }
}

// ================= main fused edge-encoder kernel =================
template <bool FULL>
__global__ __launch_bounds__(128) void edge_encoder_kernel(
    int blk0,
    const float* __restrict__ arr, const int* __restrict__ ei,
    const float* __restrict__ eattr, const int* __restrict__ batchv,
    const unsigned short* __restrict__ w1b, const unsigned short* __restrict__ w2b,
    const unsigned short* __restrict__ w3b, const unsigned short* __restrict__ pwb,
    const float* __restrict__ b1, const float* __restrict__ g1, const float* __restrict__ be1,
    const float* __restrict__ b2, const float* __restrict__ g2, const float* __restrict__ be2,
    const float* __restrict__ b3, const float* __restrict__ g3, const float* __restrict__ be3,
    const float* __restrict__ pb,
    float* __restrict__ distb, float* __restrict__ deg,
    float* __restrict__ cnt_s, float* __restrict__ cnt_d,
    float* __restrict__ sum_s, float* __restrict__ sum_d,
    float* __restrict__ eg_sum, float* __restrict__ eg_cnt) {
  __shared__ alignas(16) unsigned short sA[2][64 * 32];   // 2 x 4KB  A staging
  __shared__ alignas(16) unsigned short sB[2][32 * HB];   // 2 x 8KB  B staging
  __shared__ alignas(16) unsigned short sH[64 * HB];      // 16KB activations
  __shared__ int   lsrc[64], ldst[64], lgb[64];
  __shared__ float lext[64][4];

  const int tid = threadIdx.x;
  const int lane = tid & 31;
  const int wv = tid >> 5;      // 4 waves, 16 edges each
  const int e0 = (blk0 + blockIdx.x) * 64;

  // --- phase A: geometry features + cheap per-edge atomics ---
  if (tid < 64) {
    int e = e0 + tid;
    if (FULL || e < NED) {
      int s = ei[e], d = ei[NED + e];
      float rx = arr[3 * s + 0] - arr[3 * d + 0];
      float ry = arr[3 * s + 1] - arr[3 * d + 1];
      float rz = arr[3 * s + 2] - arr[3 * d + 2];
      float dist = sqrtf(rx * rx + ry * ry + rz * rz);
      float inv = 1.0f / (dist + 1e-8f);
      lext[tid][0] = rx * inv; lext[tid][1] = ry * inv;
      lext[tid][2] = rz * inv; lext[tid][3] = logf(dist + 1e-6f);
      int gb = batchv[s];
      lsrc[tid] = s; ldst[tid] = d; lgb[tid] = gb;
      distb[e] = dist;
      atomicAdd(deg + d, dist);
      atomicAdd(cnt_s + s, 1.0f);
      atomicAdd(cnt_d + d, 1.0f);
      atomicAdd(eg_cnt + gb, 1.0f);
    } else {
      lsrc[tid] = -1; ldst[tid] = -1; lgb[tid] = -1;
      lext[tid][0] = lext[tid][1] = lext[tid][2] = lext[tid][3] = 0.f;
    }
  }
  __syncthreads();

  v8f acc[8];
  const v8f zz = {0.f, 0.f, 0.f, 0.f, 0.f, 0.f, 0.f, 0.f};
#pragma unroll
  for (int t = 0; t < 8; ++t) acc[t] = zz;

  // --- layer 1: (64 x 2080) @ (2080 x 128), software-pipelined, 1 barrier/chunk ---
  stageA_main<FULL>(sA[0], eattr, e0, 0, tid);
  stageB(w1b, sB[0], tid);
  for (int kc = 0; kc < NKC; ++kc) {
    __syncthreads();                         // chunk kc staged everywhere
    int cur = kc & 1, nxt = cur ^ 1;
    if (kc + 1 < NKC) {
      if (kc + 1 < NKC - 1)
        stageA_main<FULL>(sA[nxt], eattr, e0, kc + 1, tid);
      else
        stageA_last(sA[nxt], eattr, lext, e0, tid);
      stageB(w1b + (size_t)(kc + 1) * 32 * HB, sB[nxt], tid);
    }
    BF16x16 a = load_afrag(sA[cur] + wv * 16 * 32, lane, 32);
    wmma_8tiles(acc, a, sB[cur], lane);
  }
  add_bias_relu(acc, b1, lane);
  ln_rows(acc, g1, be1, lane);
  store_sH(sH, acc, wv, lane);

  // --- layers 2, 3 ---
  mm_sH(acc, sH, w2b, sB, wv, lane, tid);
  add_bias_relu(acc, b2, lane);
  ln_rows(acc, g2, be2, lane);
  store_sH(sH, acc, wv, lane);

  mm_sH(acc, sH, w3b, sB, wv, lane, tid);
  add_bias_relu(acc, b3, lane);
  ln_rows(acc, g3, be3, lane);       // acc == edge_emb

  // edge-graph pooling of edge_emb
  {
    const int kh = lane >> 4, c0 = lane & 15;
#pragma unroll
    for (int r = 0; r < 8; ++r) {
      int el = wv * 16 + r + kh * 8;
      int gb = lgb[el];
      if (gb >= 0) {
#pragma unroll
        for (int t = 0; t < 8; ++t)
          atomicAdd(eg_sum + gb * HB + t * 16 + c0, acc[t][r]);
      }
    }
  }
  store_sH(sH, acc, wv, lane);

  // --- projection: p = relu(edge_emb @ proj_w + proj_b) ---
  mm_sH(acc, sH, pwb, sB, wv, lane, tid);
  add_bias_relu(acc, pb, lane);

  // segment sums of p by src and dst
  {
    const int kh = lane >> 4, c0 = lane & 15;
#pragma unroll
    for (int r = 0; r < 8; ++r) {
      int el = wv * 16 + r + kh * 8;
      int s = lsrc[el];
      if (s >= 0) {
        int d = ldst[el];
#pragma unroll
        for (int t = 0; t < 8; ++t) {
          float v = acc[t][r];
          int c = t * 16 + c0;
          atomicAdd(sum_s + (size_t)s * HB + c, v);
          atomicAdd(sum_d + (size_t)d * HB + c, v);
        }
      }
    }
  }
}

// ================= node-feature WMMA matmul: h = x @ W =================
template <bool FULL>
__global__ __launch_bounds__(128) void node_gemm_kernel(
    int blk0, float* __restrict__ hout, const float* __restrict__ xin,
    const unsigned short* __restrict__ wb) {
  __shared__ alignas(16) unsigned short sA[2][64 * 32];
  __shared__ alignas(16) unsigned short sB[2][32 * HB];
  const int tid = threadIdx.x, lane = tid & 31, wv = tid >> 5;
  const int n0 = (blk0 + blockIdx.x) * 64;
  v8f acc[8];
  const v8f zz = {0.f, 0.f, 0.f, 0.f, 0.f, 0.f, 0.f, 0.f};
#pragma unroll
  for (int t = 0; t < 8; ++t) acc[t] = zz;

  auto stageA = [&](int kc, unsigned short* dst) {
#pragma unroll
    for (int i = 0; i < 8; ++i) {
      int p = tid + i * 128;
      int el = p >> 4;
      int k0 = (p & 15) * 2;
      int row = n0 + el;
      float2 v = make_float2(0.f, 0.f);
      if (FULL || row < NND)
        v = *reinterpret_cast<const float2*>(xin + (size_t)row * HB + kc * 32 + k0);
      *reinterpret_cast<unsigned*>(dst + el * 32 + k0) = pk2bf(v.x, v.y);
    }
  };

  stageA(0, sA[0]);
  stageB(wb, sB[0], tid);
#pragma unroll
  for (int kc = 0; kc < 4; ++kc) {
    __syncthreads();
    int cur = kc & 1, nxt = cur ^ 1;
    if (kc < 3) {
      stageA(kc + 1, sA[nxt]);
      stageB(wb + (size_t)(kc + 1) * 32 * HB, sB[nxt], tid);
    }
    BF16x16 a = load_afrag(sA[cur] + wv * 16 * 32, lane, 32);
    wmma_8tiles(acc, a, sB[cur], lane);
  }
  const int kh = lane >> 4, c0 = lane & 15;
#pragma unroll
  for (int r = 0; r < 8; ++r) {
    int row = n0 + wv * 16 + r + kh * 8;
    if (FULL || row < NND) {
#pragma unroll
      for (int t = 0; t < 8; ++t)
        hout[(size_t)row * HB + t * 16 + c0] = acc[t][r];
    }
  }
}

// ================= small helper kernels =================
__global__ void fill_kernel(float* p, float v, size_t n) {
  size_t i = (size_t)blockIdx.x * blockDim.x + threadIdx.x;
  if (i < n) p[i] = v;
}

__global__ void convW1_kernel(unsigned short* dst, const float* __restrict__ w) {
  int idx = blockIdx.x * blockDim.x + threadIdx.x;
  if (idx < KPAD * HB) {
    int k = idx >> 7, j = idx & 127;
    float v = (k < FDIM) ? w[(size_t)k * HB + j] : 0.f;
    dst[idx] = (unsigned short)f2bf(v);
  }
}

__global__ void convW_kernel(unsigned short* dst, const float* __restrict__ w, int n) {
  int idx = blockIdx.x * blockDim.x + threadIdx.x;
  if (idx < n) dst[idx] = (unsigned short)f2bf(w[idx]);
}

__global__ void combine_kernel(float* __restrict__ x,
                               const float* __restrict__ sum_s, const float* __restrict__ sum_d,
                               const float* __restrict__ cnt_s, const float* __restrict__ cnt_d) {
  int idx = blockIdx.x * blockDim.x + threadIdx.x;
  if (idx < NND * HB) {
    int v = idx >> 7;
    x[idx] = 0.5f * (sum_s[idx] / fmaxf(cnt_s[v], 1.f) +
                     sum_d[idx] / fmaxf(cnt_d[v], 1.f));
  }
}

__global__ void dinv_kernel(float* __restrict__ dinvb, const float* __restrict__ deg) {
  int i = blockIdx.x * blockDim.x + threadIdx.x;
  if (i < NND) {
    float d = deg[i];
    dinvb[i] = d > 0.f ? rsqrtf(fmaxf(d, 1e-30f)) : 0.f;
  }
}

__global__ void selfloop_kernel(float* __restrict__ accb, const float* __restrict__ hbuf,
                                const float* __restrict__ dinvb) {
  int idx = blockIdx.x * blockDim.x + threadIdx.x;
  if (idx < NND * HB) {
    int v = idx >> 7;
    float di = dinvb[v];
    accb[idx] = hbuf[idx] * di * di;
  }
}

__global__ void scatter_kernel(float* __restrict__ accb, const float* __restrict__ hbuf,
                               const float* __restrict__ dinvb, const int* __restrict__ ei,
                               const float* __restrict__ distb) {
  int idx = blockIdx.x * blockDim.x + threadIdx.x;
  if (idx < NED * HB) {
    int e = idx >> 7, j = idx & 127;
    int s = ei[e], d = ei[NED + e];
    float nrm = dinvb[s] * distb[e] * dinvb[d];
    atomicAdd(accb + (size_t)d * HB + j, hbuf[(size_t)s * HB + j] * nrm);
  }
}

// x := LN(accb + b + x) [optionally exact GELU]; one row per 32-lane wave
__global__ __launch_bounds__(256) void postnorm_kernel(
    float* __restrict__ x, const float* __restrict__ accb,
    const float* __restrict__ b, const float* __restrict__ g,
    const float* __restrict__ be, int do_gelu) {
  int lane = threadIdx.x & 31, wvi = threadIdx.x >> 5;
  int row = blockIdx.x * 8 + wvi;
  if (row >= NND) return;
  float v[4];
#pragma unroll
  for (int i = 0; i < 4; ++i) {
    int c = lane + i * 32;
    v[i] = accb[(size_t)row * HB + c] + b[c] + x[(size_t)row * HB + c];
  }
  float s = v[0] + v[1] + v[2] + v[3];
  for (int m = 1; m < 32; m <<= 1) s += __shfl_xor(s, m, 32);
  float mean = s * (1.0f / 128.0f);
  float s2 = 0.f;
#pragma unroll
  for (int i = 0; i < 4; ++i) { float d = v[i] - mean; s2 += d * d; }
  for (int m = 1; m < 32; m <<= 1) s2 += __shfl_xor(s2, m, 32);
  float inv = rsqrtf(s2 * (1.0f / 128.0f) + 1e-5f);
#pragma unroll
  for (int i = 0; i < 4; ++i) {
    int c = lane + i * 32;
    float y = (v[i] - mean) * inv * g[c] + be[c];
    if (do_gelu) y = 0.5f * y * (1.0f + erff(y * 0.70710678118f));
    x[(size_t)row * HB + c] = y;
  }
}

__global__ void pool_kernel(float* __restrict__ ng_sum, float* __restrict__ ng_cnt,
                            const float* __restrict__ x, const int* __restrict__ batchv) {
  int idx = blockIdx.x * blockDim.x + threadIdx.x;
  if (idx < NND * HB) {
    int v = idx >> 7, j = idx & 127;
    int gb = batchv[v];
    atomicAdd(ng_sum + gb * HB + j, x[idx]);
    if (j == 0) atomicAdd(ng_cnt + gb, 1.0f);
  }
}

__global__ __launch_bounds__(256) void final_kernel(
    float* __restrict__ out,
    const float* __restrict__ ng_sum, const float* __restrict__ ng_cnt,
    const float* __restrict__ eg_sum, const float* __restrict__ eg_cnt,
    const float* __restrict__ h1w, const float* __restrict__ h1b,
    const float* __restrict__ h2w, const float* __restrict__ h2b) {
  __shared__ float gm[16 * 256];
  __shared__ float t1[16 * 128];
  int tid = threadIdx.x;
  for (int idx = tid; idx < 16 * 256; idx += 256) {
    int i = idx >> 8, c = idx & 255;
    float v = (c < 128) ? ng_sum[i * HB + c] / fmaxf(ng_cnt[i], 1.f)
                        : eg_sum[i * HB + (c - 128)] / fmaxf(eg_cnt[i], 1.f);
    gm[idx] = v;
  }
  __syncthreads();
  for (int idx = tid; idx < 16 * 128; idx += 256) {
    int i = idx >> 7, j = idx & 127;
    float s = h1b[j];
    for (int k = 0; k < 256; ++k) s += gm[i * 256 + k] * h1w[k * 128 + j];
    t1[idx] = s > 0.f ? s : 0.f;
  }
  __syncthreads();
  if (tid < 32) {
    int i = tid >> 1, o = tid & 1;
    float s = h2b[o];
    for (int j = 0; j < 128; ++j) s += t1[i * 128 + j] * h2w[j * 2 + o];
    out[i * 2 + o] = s;
  }
}

// ================= host launch =================
extern "C" void kernel_launch(void* const* d_in, const int* in_sizes, int n_in,
                              void* d_out, int out_size, void* d_ws, size_t ws_size,
                              hipStream_t stream) {
  const float* arr    = (const float*)d_in[0];
  const int*   ei     = (const int*)d_in[1];
  const float* eattr  = (const float*)d_in[2];
  const int*   batchv = (const int*)d_in[3];
  const float* enc_w1 = (const float*)d_in[4];
  const float* enc_b1 = (const float*)d_in[5];
  const float* enc_g1 = (const float*)d_in[6];
  const float* enc_be1 = (const float*)d_in[7];
  const float* enc_w2 = (const float*)d_in[8];
  const float* enc_b2 = (const float*)d_in[9];
  const float* enc_g2 = (const float*)d_in[10];
  const float* enc_be2 = (const float*)d_in[11];
  const float* enc_w3 = (const float*)d_in[12];
  const float* enc_b3 = (const float*)d_in[13];
  const float* enc_g3 = (const float*)d_in[14];
  const float* enc_be3 = (const float*)d_in[15];
  const float* proj_w = (const float*)d_in[16];
  const float* proj_b = (const float*)d_in[17];
  const float* g1_w = (const float*)d_in[18];
  const float* g1_b = (const float*)d_in[19];
  const float* g2_w = (const float*)d_in[20];
  const float* g2_b = (const float*)d_in[21];
  const float* n1_g = (const float*)d_in[22];
  const float* n1_b = (const float*)d_in[23];
  const float* n2_g = (const float*)d_in[24];
  const float* n2_b = (const float*)d_in[25];
  const float* h1_w = (const float*)d_in[26];
  const float* h1_b = (const float*)d_in[27];
  const float* h2_w = (const float*)d_in[28];
  const float* h2_b = (const float*)d_in[29];
  (void)in_sizes; (void)n_in; (void)out_size; (void)ws_size;

  // ---- carve workspace (all 256B-aligned) ----
  char* ws = (char*)d_ws;
  size_t off = 0;
  auto take = [&](size_t bytes) -> void* {
    void* p = ws + off;
    off += (bytes + 255) & ~(size_t)255;
    return p;
  };
  unsigned short* w1b = (unsigned short*)take((size_t)KPAD * HB * 2);
  unsigned short* w2b = (unsigned short*)take((size_t)HB * HB * 2);
  unsigned short* w3b = (unsigned short*)take((size_t)HB * HB * 2);
  unsigned short* pwb = (unsigned short*)take((size_t)HB * HB * 2);
  unsigned short* g1b = (unsigned short*)take((size_t)HB * HB * 2);
  unsigned short* g2b = (unsigned short*)take((size_t)HB * HB * 2);
  float* distb = (float*)take((size_t)NED * 4);
  float* deg   = (float*)take((size_t)NND * 4);
  float* dinvb = (float*)take((size_t)NND * 4);
  // contiguous zero-init region
  const size_t ZTOTAL = (size_t)2 * NND + (size_t)2 * NND * HB + 2 * (NGR * HB + NGR);
  float* zbase = (float*)take(ZTOTAL * 4);
  float* cnt_s = zbase;
  float* cnt_d = cnt_s + NND;
  float* sum_s = cnt_d + NND;
  float* sum_d = sum_s + (size_t)NND * HB;
  float* eg_sum = sum_d + (size_t)NND * HB;
  float* eg_cnt = eg_sum + NGR * HB;
  float* ng_sum = eg_cnt + NGR;
  float* ng_cnt = ng_sum + NGR * HB;
  float* xbuf = (float*)take((size_t)NND * HB * 4);
  // reuse segment-sum buffers after they are consumed
  float* hbuf = sum_s;
  float* accb = sum_d;

  // ---- init ----
  fill_kernel<<<(NND + 255) / 256, 256, 0, stream>>>(deg, 1.0f, (size_t)NND); // self-loop
  fill_kernel<<<(unsigned)((ZTOTAL + 255) / 256), 256, 0, stream>>>(zbase, 0.0f, ZTOTAL);

  // ---- weight conversion to bf16 ----
  convW1_kernel<<<(KPAD * HB + 255) / 256, 256, 0, stream>>>(w1b, enc_w1);
  convW_kernel<<<64, 256, 0, stream>>>(w2b, enc_w2, HB * HB);
  convW_kernel<<<64, 256, 0, stream>>>(w3b, enc_w3, HB * HB);
  convW_kernel<<<64, 256, 0, stream>>>(pwb, proj_w, HB * HB);
  convW_kernel<<<64, 256, 0, stream>>>(g1b, g1_w, HB * HB);
  convW_kernel<<<64, 256, 0, stream>>>(g2b, g2_w, HB * HB);

  // ---- fused edge encoder + projection + segment sums ----
  edge_encoder_kernel<true><<<EFULL, 128, 0, stream>>>(
      0, arr, ei, eattr, batchv, w1b, w2b, w3b, pwb,
      enc_b1, enc_g1, enc_be1, enc_b2, enc_g2, enc_be2, enc_b3, enc_g3, enc_be3,
      proj_b, distb, deg, cnt_s, cnt_d, sum_s, sum_d, eg_sum, eg_cnt);
  edge_encoder_kernel<false><<<1, 128, 0, stream>>>(
      EFULL, arr, ei, eattr, batchv, w1b, w2b, w3b, pwb,
      enc_b1, enc_g1, enc_be1, enc_b2, enc_g2, enc_be2, enc_b3, enc_g3, enc_be3,
      proj_b, distb, deg, cnt_s, cnt_d, sum_s, sum_d, eg_sum, eg_cnt);

  // ---- x = 0.5*(mean_src + mean_dst) ----
  combine_kernel<<<(NND * HB) / 256, 256, 0, stream>>>(xbuf, sum_s, sum_d, cnt_s, cnt_d);
  dinv_kernel<<<(NND + 255) / 256, 256, 0, stream>>>(dinvb, deg);

  // ---- GCN conv 1 ----
  node_gemm_kernel<true><<<NFULL, 128, 0, stream>>>(0, hbuf, xbuf, g1b);
  node_gemm_kernel<false><<<1, 128, 0, stream>>>(NFULL, hbuf, xbuf, g1b);
  selfloop_kernel<<<(NND * HB) / 256, 256, 0, stream>>>(accb, hbuf, dinvb);
  scatter_kernel<<<(NED * HB) / 256, 256, 0, stream>>>(accb, hbuf, dinvb, ei, distb);
  postnorm_kernel<<<(NND + 7) / 8, 256, 0, stream>>>(xbuf, accb, g1_b, n1_g, n1_b, 1);

  // ---- GCN conv 2 ----
  node_gemm_kernel<true><<<NFULL, 128, 0, stream>>>(0, hbuf, xbuf, g2b);
  node_gemm_kernel<false><<<1, 128, 0, stream>>>(NFULL, hbuf, xbuf, g2b);
  selfloop_kernel<<<(NND * HB) / 256, 256, 0, stream>>>(accb, hbuf, dinvb);
  scatter_kernel<<<(NED * HB) / 256, 256, 0, stream>>>(accb, hbuf, dinvb, ei, distb);
  postnorm_kernel<<<(NND + 7) / 8, 256, 0, stream>>>(xbuf, accb, g2_b, n2_g, n2_b, 0);

  // ---- pooling + head MLP ----
  pool_kernel<<<(NND * HB) / 256, 256, 0, stream>>>(ng_sum, ng_cnt, xbuf, batchv);
  final_kernel<<<1, 256, 0, stream>>>((float*)d_out, ng_sum, ng_cnt, eg_sum, eg_cnt,
                                      h1_w, h1_b, h2_w, h2_b);
}
